// VectorQuantizerEMA_55671366090817
// MI455X (gfx1250) — matrix operand order repaired and checked by reading, not compiled, not run on hardware
//
#include <hip/hip_runtime.h>
#include <hip/hip_bf16.h>

typedef __attribute__((ext_vector_type(16))) __bf16 v16bf;
typedef __attribute__((ext_vector_type(8)))  float  v8f;
typedef __attribute__((ext_vector_type(4)))  float  v4f;

#define N_ROWS 32768
#define K_CODES 8192
#define D_DIM 256
#define DECAY 0.99f
#define ONE_M_DECAY 0.01f
#define BETA 0.25f
#define EPS 1e-5f
#define FIXSCALE 1048576.0f      // 2^20 fixed point for deterministic segment sums
#define INV_FIXSCALE (1.0f/1048576.0f)

#define NTILES (K_CODES / 16)          // 512 column tiles
#define TILE_BF16 8192                 // bf16 elements per packed tile (hi 4096 + lo 4096)
#define TILE_BYTES 16384               // 16 KB per tile (hi 8KB + lo 8KB)

// ---------------------------------------------------------------------------
// Kernel 1: split codebook into bf16 hi/lo, packed per 16-column tile in the
// WMMA B-operand lane layout, + ||w||^2.
// Tile ct occupies TILE_BF16 elements: hi at [0,4096), lo at [4096,8192).
// Within-half index: ((c*2 + half)*16 + n)*16 + e  where
//   k = ct*16 + n (column), d = c*32 + half*16 + e (contraction dim).
// B k order chosen here (A uses the same): lane L -> col n=L%16, half=L/16;
// lane's 16 contiguous bf16 cover K = 16*half + e.
// ---------------------------------------------------------------------------
__global__ void pack_codebook(const float* __restrict__ emb,
                              __bf16* __restrict__ wpk,
                              float* __restrict__ wnorm) {
    const int k = blockIdx.x;          // code index
    const int d = threadIdx.x;         // 256 threads = D
    const float w = emb[(size_t)k * D_DIM + d];
    const __bf16 h = (__bf16)w;
    const __bf16 l = (__bf16)(w - (float)h);
    const int ct = k >> 4, n = k & 15;
    const int c = d >> 5, rem = d & 31, hf = rem >> 4, e = rem & 15;
    const size_t base = (size_t)ct * TILE_BF16 + (((c * 2 + hf) * 16 + n) * 16 + e);
    wpk[base] = h;
    wpk[base + 4096] = l;
    __shared__ float red[256];
    red[d] = w * w;
    __syncthreads();
    for (int s = 128; s > 0; s >>= 1) {
        if (d < s) red[d] += red[d + s];
        __syncthreads();
    }
    if (d == 0) wnorm[k] = red[0];
}

// ---------------------------------------------------------------------------
// Kernel 2: zero the integer accumulators (cnt u32 [K] followed by avg i64 [K*D])
// ---------------------------------------------------------------------------
__global__ void zero_accum(unsigned* __restrict__ p, size_t n_u32) {
    size_t i = (size_t)blockIdx.x * blockDim.x + threadIdx.x;
    if (i < n_u32) p[i] = 0u;
}

// ---------------------------------------------------------------------------
// Kernel 3: fused distance GEMM (split-bf16, v_wmma_f32_16x16x32_bf16) + argmin.
// Grid: N/128 blocks of 256 threads (8 waves). Each wave owns 16 private rows
// with its A fragments resident in VGPRs. All 8 waves share each 16 KB B tile,
// double-buffered through LDS (global_load_b128 -> compute -> ds_store -> barrier).
// Three independent accumulator chains (hi*hi, hi*lo, lo*hi) for XDL ILP.
// Score = ||w||^2 - 2*(z.w)   (||z||^2 is row-constant, irrelevant for argmin).
// ---------------------------------------------------------------------------
__global__ void __launch_bounds__(256)
argmin_wmma(const float* __restrict__ z_e,
            const __bf16* __restrict__ wpk,
            const float* __restrict__ wnorm,
            int* __restrict__ widx, float* __restrict__ out_idx) {
    const int tid  = threadIdx.x;
    const int lane = tid & 31;
    const int wave = tid >> 5;         // 0..7
    const int half = lane >> 4;
    const int n    = lane & 15;

    __shared__ __bf16 sB[2][TILE_BF16];        // 2 x 16 KB double buffer
    __shared__ float  sBest[128 * 16];         // [wave*16+row][slot]
    __shared__ int    sIdx[128 * 16];

    // ---- preload this wave's A (16 rows x D) as hi/lo bf16 fragments --------
    // A k order chosen here (B packed to match): lane holds row m = lane%16; element e covers
    // K(e) = 16*half + e within the 32-wide chunk.
    const int row = blockIdx.x * 128 + wave * 16 + n;
    const float* zrow = z_e + (size_t)row * D_DIM;
    v16bf a_hi[8], a_lo[8];
#pragma unroll
    for (int c = 0; c < 8; ++c) {
        const float* g0 = zrow + 32 * c + 16 * half;      // e = 0..7
        const float* g1 = zrow + 32 * c + 16 * half + 8;  // e = 8..15
        v4f t0 = *(const v4f*)(g0);
        v4f t1 = *(const v4f*)(g0 + 4);
        v4f t2 = *(const v4f*)(g1);
        v4f t3 = *(const v4f*)(g1 + 4);
        float f[16];
#pragma unroll
        for (int i = 0; i < 4; ++i) {
            f[i] = t0[i]; f[4 + i] = t1[i]; f[8 + i] = t2[i]; f[12 + i] = t3[i];
        }
#pragma unroll
        for (int e = 0; e < 16; ++e) {
            float x = f[e];
            __bf16 h = (__bf16)x;
            a_hi[c][e] = h;
            a_lo[c][e] = (__bf16)(x - (float)h);
        }
    }

    // ---- stage tile 0 into buffer 0 (16 KB, 64B per thread) -----------------
    {
        const uint4* gsrc = (const uint4*)wpk;   // tile 0
        uint4* ldst = (uint4*)sB[0];
#pragma unroll
        for (int i = 0; i < 4; ++i) ldst[tid + 256 * i] = gsrc[tid + 256 * i];
    }
    __syncthreads();

    // ---- sweep column tiles ------------------------------------------------
    float best[8];
    int   bidx[8];
#pragma unroll
    for (int j = 0; j < 8; ++j) { best[j] = 3.4e38f; bidx[j] = 0x7fffffff; }

    for (int t = 0; t < NTILES; ++t) {
        const int cb = t & 1, nb = cb ^ 1;
        const bool hasNext = (t + 1) < NTILES;

        // issue next tile's global loads first (latency hides under WMMAs)
        uint4 st0, st1, st2, st3;
        if (hasNext) {
            const uint4* gsrc = (const uint4*)(wpk + (size_t)(t + 1) * TILE_BF16);
            st0 = gsrc[tid];
            st1 = gsrc[tid + 256];
            st2 = gsrc[tid + 512];
            st3 = gsrc[tid + 768];
        }

        // 24 WMMAs on current buffer; 3 independent accumulator chains
        v8f acc0 = {0.f, 0.f, 0.f, 0.f, 0.f, 0.f, 0.f, 0.f};
        v8f acc1 = acc0, acc2 = acc0;
#pragma unroll
        for (int c = 0; c < 8; ++c) {
            const __bf16* bp = &sB[cb][((c * 2 + half) * 16 + n) * 16];
            const v16bf bh = *(const v16bf*)bp;            // ds_load 32B
            const v16bf bl = *(const v16bf*)(bp + 4096);
            // z.w ~= zh*wh + zh*wl + zl*wh  (zl*wl below fp32 noise floor)
            acc0 = __builtin_amdgcn_wmma_f32_16x16x32_bf16(
                     false, a_hi[c], false, bh, (short)0, acc0, false, false);
            acc1 = __builtin_amdgcn_wmma_f32_16x16x32_bf16(
                     false, a_hi[c], false, bl, (short)0, acc1, false, false);
            acc2 = __builtin_amdgcn_wmma_f32_16x16x32_bf16(
                     false, a_lo[c], false, bh, (short)0, acc2, false, false);
        }

        // argmin epilogue: lane holds col n of this tile, rows m = j + 8*half
        const int col = t * 16 + n;
        const float wn = wnorm[col];
#pragma unroll
        for (int j = 0; j < 8; ++j) {
            float s = wn - 2.0f * (acc0[j] + acc1[j] + acc2[j]);
            if (s < best[j] || (s == best[j] && col < bidx[j])) {
                best[j] = s; bidx[j] = col;
            }
        }

        // commit next tile into the other buffer, then block barrier
        if (hasNext) {
            uint4* ldst = (uint4*)sB[nb];
            ldst[tid]       = st0;
            ldst[tid + 256] = st1;
            ldst[tid + 512] = st2;
            ldst[tid + 768] = st3;
        }
        __syncthreads();
    }

    // ---- per-(wave,row) argmin reduction across the 16 lane-columns ---------
#pragma unroll
    for (int j = 0; j < 8; ++j) {
        const int m = j + 8 * half;                    // row within wave tile
        sBest[(wave * 16 + m) * 16 + n] = best[j];
        sIdx[(wave * 16 + m) * 16 + n]  = bidx[j];
    }
    __syncthreads();
    if (tid < 128) {                                   // one thread per block row
        const int base = tid * 16;
        float b = sBest[base];
        int bi = sIdx[base];
        for (int s = 1; s < 16; ++s) {
            float v = sBest[base + s];
            int vi = sIdx[base + s];
            if (v < b || (v == b && vi < bi)) { b = v; bi = vi; }
        }
        const int grow = blockIdx.x * 128 + tid;
        widx[grow] = bi;
        out_idx[grow] = (float)bi;
    }
}

// ---------------------------------------------------------------------------
// Kernel 4: gather z_q, commitment partials, deterministic segment sums
// (u32 count atomics; s20.20 fixed-point i64 atomics for avg => bit-determinism)
// ---------------------------------------------------------------------------
__global__ void gather_scatter(const float* __restrict__ z_e,
                               const float* __restrict__ emb,
                               const int* __restrict__ widx,
                               float* __restrict__ out_zq,
                               unsigned* __restrict__ cnt,
                               unsigned long long* __restrict__ avg,
                               float* __restrict__ commit_partial) {
    const int row = blockIdx.x;
    const int d = threadIdx.x;
    const int k = widx[row];
    const float z = z_e[(size_t)row * D_DIM + d];
    const float q = emb[(size_t)k * D_DIM + d];
    out_zq[(size_t)row * D_DIM + d] = q;   // z_q_st == z_q numerically
    const float diff = z - q;
    const long long s = llrintf(z * FIXSCALE);
    atomicAdd(&avg[(size_t)k * D_DIM + d], (unsigned long long)s);
    __shared__ float red[256];
    red[d] = diff * diff;
    __syncthreads();
    for (int s2 = 128; s2 > 0; s2 >>= 1) {
        if (d < s2) red[d] += red[d + s2];
        __syncthreads();
    }
    if (d == 0) {
        commit_partial[row] = red[0];
        atomicAdd(&cnt[k], 1u);
    }
}

// ---------------------------------------------------------------------------
// Kernel 5: single block, fixed-order reductions: new_count, n, commitment
// ---------------------------------------------------------------------------
__global__ void reduce_stats(const float* __restrict__ ema_count,
                             const unsigned* __restrict__ cnt,
                             float* __restrict__ out_newcount,
                             float* __restrict__ out_commit,
                             const float* __restrict__ commit_partial,
                             float* __restrict__ nscalar) {
    __shared__ float red[256];
    const int t = threadIdx.x;
    float local = 0.f;
    for (int k = t; k < K_CODES; k += 256) {
        float nc = DECAY * ema_count[k] + ONE_M_DECAY * (float)cnt[k];
        out_newcount[k] = nc;
        local += nc;
    }
    red[t] = local;
    __syncthreads();
    for (int s = 128; s > 0; s >>= 1) {
        if (t < s) red[t] += red[t + s];
        __syncthreads();
    }
    if (t == 0) nscalar[0] = red[0];
    __syncthreads();
    float lc = 0.f;
    for (int i = t; i < N_ROWS; i += 256) lc += commit_partial[i];
    red[t] = lc;
    __syncthreads();
    for (int s = 128; s > 0; s >>= 1) {
        if (t < s) red[t] += red[t + s];
        __syncthreads();
    }
    if (t == 0) out_commit[0] = BETA * red[0] / ((float)N_ROWS * (float)D_DIM);
}

// ---------------------------------------------------------------------------
// Kernel 6: new_avg = DECAY*ema_avg + (1-DECAY)*seg_sum ; new_embed = new_avg/cs
// ---------------------------------------------------------------------------
__global__ void finalize(const float* __restrict__ ema_avg,
                         const unsigned long long* __restrict__ avg,
                         const float* __restrict__ newcount,
                         const float* __restrict__ nscalar,
                         float* __restrict__ out_newavg,
                         float* __restrict__ out_newembed) {
    const size_t i = (size_t)blockIdx.x * 256 + threadIdx.x;  // over K*D
    const int k = (int)(i >> 8);
    const float a = (float)(long long)avg[i] * INV_FIXSCALE;
    const float na = DECAY * ema_avg[i] + ONE_M_DECAY * a;
    out_newavg[i] = na;
    const float nn = nscalar[0];
    const float cs = (newcount[k] + EPS) / (nn + (float)K_CODES * EPS) * nn;
    out_newembed[i] = na / cs;
}

// ---------------------------------------------------------------------------
extern "C" void kernel_launch(void* const* d_in, const int* in_sizes, int n_in,
                              void* d_out, int out_size, void* d_ws, size_t ws_size,
                              hipStream_t stream) {
    const float* z_e       = (const float*)d_in[0];   // [N, D]
    const float* embedding = (const float*)d_in[1];   // [K, D]
    const float* ema_count = (const float*)d_in[2];   // [K]
    const float* ema_avg   = (const float*)d_in[3];   // [K, D]

    // output layout (floats, concatenated in return order)
    float* out = (float*)d_out;
    const size_t o_zq       = 0;
    const size_t o_commit   = (size_t)N_ROWS * D_DIM;                 // 8388608
    const size_t o_idx      = o_commit + 1;
    const size_t o_newembed = o_idx + N_ROWS;
    const size_t o_newcount = o_newembed + (size_t)K_CODES * D_DIM;
    const size_t o_newavg   = o_newcount + K_CODES;

    // workspace carve-up (256B-aligned regions; cnt and avg kept contiguous)
    char* ws = (char*)d_ws;
    size_t off = 0;
    auto carve = [&](size_t bytes) -> char* {
        char* p = ws + off;
        off = (off + bytes + 255) & ~(size_t)255;
        return p;
    };
    __bf16*             wpk     = (__bf16*)carve((size_t)NTILES * TILE_BYTES);
    float*              wnorm   = (float*)carve((size_t)K_CODES * 4);
    int*                widx    = (int*)carve((size_t)N_ROWS * 4);
    unsigned*           cnt     = (unsigned*)carve((size_t)K_CODES * 4);
    unsigned long long* avg     = (unsigned long long*)carve((size_t)K_CODES * D_DIM * 8);
    float*              commitp = (float*)carve((size_t)N_ROWS * 4);
    float*              nscal   = (float*)carve(256);
    if (off > ws_size) return;  // insufficient scratch; bail safely

    // 1) pack codebook into tile-major WMMA-layout bf16 hi/lo + ||w||^2
    pack_codebook<<<K_CODES, 256, 0, stream>>>(embedding, wpk, wnorm);

    // 2) zero integer accumulators (cnt + avg are contiguous u32 span)
    {
        size_t n_u32 = (size_t)K_CODES + (size_t)K_CODES * D_DIM * 2;
        unsigned blocks = (unsigned)((n_u32 + 255) / 256);
        zero_accum<<<blocks, 256, 0, stream>>>(cnt, n_u32);
    }

    // 3) fused split-bf16 WMMA distance GEMM + argmin (LDS double-buffered B)
    argmin_wmma<<<N_ROWS / 128, 256, 0, stream>>>(z_e, wpk, wnorm,
                                                  widx, out + o_idx);

    // 4) gather + commitment partials + deterministic segment sums
    gather_scatter<<<N_ROWS, 256, 0, stream>>>(z_e, embedding, widx,
                                               out + o_zq, cnt, avg, commitp);

    // 5) new_count, n, commitment (fixed-order => deterministic)
    reduce_stats<<<1, 256, 0, stream>>>(ema_count, cnt, out + o_newcount,
                                        out + o_commit, commitp, nscal);

    // 6) new_avg + new_embed
    finalize<<<(K_CODES * D_DIM) / 256, 256, 0, stream>>>(ema_avg, avg,
                                                          out + o_newcount, nscal,
                                                          out + o_newavg,
                                                          out + o_newembed);
}
